// TextEncode_21148418966049
// MI455X (gfx1250) — matrix operand order repaired and checked by reading, not compiled, not run on hardware
//
#include <hip/hip_runtime.h>

// ---------------------------------------------------------------------------
// VQ-VAE vector quantization for MI455X (gfx1250, wave32, WMMA).
//   x:   [32,1024,256] f32  -> N=32768 rows, D=256
//   E:   [256, 8192]   f32  -> K=8192 codes stored column-wise
//   out: quantized [N,256] f32  ++  loss scalar (1 float)
// Distance GEMM (137 GFLOP) via v_wmma_f32_16x16x32_bf16; argmin uses
// score = ||e||^2 - 2*x.e (||x||^2 is row-constant, irrelevant for argmin).
// Two independent accumulator chains per iteration for XDL ILP.
// Gather + loss use the exact f32 codebook.
// ---------------------------------------------------------------------------

typedef __bf16 bf16;
typedef __attribute__((ext_vector_type(16))) __bf16 v16bf;
typedef __attribute__((ext_vector_type(8)))  float  v8f;

#define D_DIM       256
#define K_CODES     8192
#define N_ROWS      32768
#define ROWS_PER_WG 128
#define WAVES_PER_WG 8
#define NTILES      (K_CODES / 16)   // 512 code tiles of 16
#define DCHUNKS     (D_DIM / 32)     // 8 WMMA k-steps of 32

// ---------------------------------------------------------------------------
// Repack f32 embedding [D][K] into bf16 B-fragments.
// Fragment (t, c): 32x16 tile, codes [16t,16t+16), D [32c,32c+32).
// CDNA5 16-bit B layout (32x16): lanes 0-15: e<8 -> K=e, e>=8 -> K=e+8;
// lanes 16-31: e<8 -> K=e+8, e>=8 -> K=e+16; column N = lane & 15.
// Storage: embB[((t*8 + c)*32 + lane)*16 + e]  -> per-lane 32B contiguous.
// ---------------------------------------------------------------------------
__global__ void vq_prep_embB(const float* __restrict__ emb, bf16* __restrict__ embB) {
    int u    = blockIdx.x * blockDim.x + threadIdx.x;   // 0 .. 2097151
    int e    = u & 15;
    int lane = (u >> 4) & 31;
    int c    = (u >> 9) & 7;
    int t    = u >> 12;
    int klocal = (e < 8) ? (e + ((lane < 16) ? 0 : 8)) : (e + ((lane < 16) ? 8 : 16));
    int code   = t * 16 + (lane & 15);
    int d      = c * 32 + klocal;
    embB[u] = (bf16)emb[(size_t)d * K_CODES + code];
}

// ||e_k||^2 in f32 (exact).
__global__ void vq_enorm(const float* __restrict__ emb, float* __restrict__ enorm) {
    int k = blockIdx.x * blockDim.x + threadIdx.x;      // 0..8191
    float acc = 0.0f;
    for (int d = 0; d < D_DIM; ++d) {
        float v = emb[(size_t)d * K_CODES + k];
        acc = fmaf(v, v, acc);
    }
    enorm[k] = acc;
}

// ---------------------------------------------------------------------------
// Main: per wave, 16 rows resident as bf16 A fragments; sweep 8192 codes with
// WMMA (2 tiles / iter, 2 accumulator chains), track argmin, gather f32
// codebook columns, accumulate loss partial.
// ---------------------------------------------------------------------------
__global__ __launch_bounds__(256)
void vq_main(const float* __restrict__ x, const float* __restrict__ emb,
             const bf16*  __restrict__ embB, const float* __restrict__ enorm,
             float* __restrict__ out, float* __restrict__ partials) {
    const int lane = threadIdx.x & 31;
    const int wave = threadIdx.x >> 5;
    const int half = lane >> 4;          // 0: lanes 0-15, 1: lanes 16-31
    const int m    = lane & 15;
    const int rowBase = blockIdx.x * ROWS_PER_WG + wave * 16;

    // ---- A fragments: 16 rows x 256 D, f32 -> bf16 in CDNA5 A layout -------
    // lanes 0-15 : M=lane;     e<8 -> K=e,    e>=8 -> K=e+8
    // lanes 16-31: M=lane-16;  e<8 -> K=e+8,  e>=8 -> K=e+16
    v16bf a[DCHUNKS];
    {
        const float* xrow = x + (size_t)(rowBase + m) * D_DIM;
#pragma unroll
        for (int c = 0; c < DCHUNKS; ++c) {
            const float* p = xrow + c * 32 + (half ? 8 : 0);
            float4 r0 = *(const float4*)(p);
            float4 r1 = *(const float4*)(p + 4);
            float4 r2 = *(const float4*)(p + 16);
            float4 r3 = *(const float4*)(p + 20);
            float f[16] = { r0.x, r0.y, r0.z, r0.w, r1.x, r1.y, r1.z, r1.w,
                            r2.x, r2.y, r2.z, r2.w, r3.x, r3.y, r3.z, r3.w };
            v16bf av;
#pragma unroll
            for (int e = 0; e < 16; ++e) av[e] = (bf16)f[e];
            a[c] = av;
        }
    }

    float minv[8];
    int   mini[8];
#pragma unroll
    for (int v = 0; v < 8; ++v) { minv[v] = 3.0e38f; mini[v] = 0; }

    // ---- sweep 512 code tiles, 2 per iteration (independent XDL chains) ----
    const v16bf* bbase = (const v16bf*)embB;
    for (int t = 0; t < NTILES; t += 2) {
        const v16bf* bp0 = bbase + (size_t)t * (DCHUNKS * 32) + lane;
        const v16bf* bp1 = bp0 + (DCHUNKS * 32);
        // prefetch 2 tiles (16 KB) ahead
        __builtin_prefetch((const void*)(bp0 + 2 * DCHUNKS * 32), 0, 3);
        __builtin_prefetch((const void*)(bp1 + 2 * DCHUNKS * 32), 0, 3);

        float en0 = enorm[t * 16 + m];
        float en1 = enorm[t * 16 + 16 + m];

        v8f acc0 = {};
        v8f acc1 = {};
#pragma unroll
        for (int c = 0; c < DCHUNKS; ++c) {
            v16bf b0 = bp0[c * 32];
            v16bf b1 = bp1[c * 32];
            acc0 = __builtin_amdgcn_wmma_f32_16x16x32_bf16(
                false, a[c], false, b0, (short)0, acc0, false, false);
            acc1 = __builtin_amdgcn_wmma_f32_16x16x32_bf16(
                false, a[c], false, b1, (short)0, acc1, false, false);
        }
        // D layout: VGPR v -> row M = v + 8*half, column N = lane&15.
        int n0 = t * 16 + m;
        int n1 = n0 + 16;
#pragma unroll
        for (int v = 0; v < 8; ++v) {
            float s0 = fmaf(acc0[v], -2.0f, en0);
            if (s0 < minv[v]) { minv[v] = s0; mini[v] = n0; }   // first-hit wins
            float s1 = fmaf(acc1[v], -2.0f, en1);
            if (s1 < minv[v]) { minv[v] = s1; mini[v] = n1; }
        }
    }

    // ---- argmin across the 16 lanes holding each row -----------------------
#pragma unroll
    for (int v = 0; v < 8; ++v) {
#pragma unroll
        for (int off = 1; off < 16; off <<= 1) {
            float ov = __shfl_xor(minv[v], off, 32);
            int   oi = __shfl_xor(mini[v], off, 32);
            if (ov < minv[v] || (ov == minv[v] && oi < mini[v])) {
                minv[v] = ov; mini[v] = oi;
            }
        }
    }

    // ---- gather exact f32 codebook columns, write out, loss partial --------
    float lossAcc = 0.0f;
#pragma unroll
    for (int r = 0; r < 16; ++r) {
        // row r lives in slot r&7; half-0 rows 0-7 (lane 0), half-1 rows 8-15 (lane 16)
        int idxr = __shfl(mini[r & 7], (r >= 8) ? 16 : 0, 32);
        size_t row = (size_t)rowBase + r;
        const float* xr = x + row * D_DIM;
        float* orow = out + row * D_DIM;
#pragma unroll
        for (int i = 0; i < 8; ++i) {
            int d = lane + 32 * i;
            float q  = emb[(size_t)d * K_CODES + idxr];  // column gather (L2-resident)
            float xv = xr[d];
            orow[d] = q;                                  // quantized_st == quantized
            float df = q - xv;
            lossAcc = fmaf(df, df, lossAcc);
        }
    }
#pragma unroll
    for (int off = 1; off < 32; off <<= 1) lossAcc += __shfl_xor(lossAcc, off, 32);
    if (lane == 0) partials[blockIdx.x * WAVES_PER_WG + wave] = lossAcc;
}

// Deterministic fixed-order loss reduction: loss = 1.25 * sum / (N*D).
__global__ void vq_loss_reduce(const float* __restrict__ partials,
                               float* __restrict__ lossOut) {
    __shared__ float s[256];
    int t = threadIdx.x;
    float acc = 0.0f;
#pragma unroll
    for (int j = 0; j < 8; ++j) acc += partials[t + 256 * j];   // 2048 partials
    s[t] = acc;
    __syncthreads();
    for (int stride = 128; stride > 0; stride >>= 1) {
        if (t < stride) s[t] += s[t + stride];
        __syncthreads();
    }
    if (t == 0) lossOut[0] = s[0] * (1.25f / 8388608.0f);
}

// ---------------------------------------------------------------------------
extern "C" void kernel_launch(void* const* d_in, const int* in_sizes, int n_in,
                              void* d_out, int out_size, void* d_ws, size_t ws_size,
                              hipStream_t stream) {
    const float* x   = (const float*)d_in[0];   // 32*1024*256
    const float* emb = (const float*)d_in[1];   // 256*8192
    float* out = (float*)d_out;                 // 8388608 quantized + 1 loss

    char*  ws       = (char*)d_ws;
    bf16*  embB     = (bf16*)ws;                                  // 4 MB
    float* enorm    = (float*)(ws + (4u << 20));                  // 32 KB
    float* partials = (float*)(ws + (4u << 20) + (32u << 10));    // 8 KB

    vq_prep_embB<<<(K_CODES * D_DIM) / 256, 256, 0, stream>>>(emb, embB);
    vq_enorm<<<K_CODES / 256, 256, 0, stream>>>(emb, enorm);
    vq_main<<<N_ROWS / ROWS_PER_WG, 256, 0, stream>>>(x, emb, embB, enorm, out, partials);
    vq_loss_reduce<<<1, 256, 0, stream>>>(partials, out + (size_t)N_ROWS * D_DIM);
}
